// Transformer_32684701122914
// MI455X (gfx1250) — matrix operand order repaired and checked by reading, not compiled, not run on hardware
//
#include <hip/hip_runtime.h>
#include <math.h>

#define Bb 4
#define Cc 768
#define Tt 512
#define Hh 12
#define Dd 64
#define NLl 6
#define FFf 3072

typedef __bf16 v16bf __attribute__((ext_vector_type(16)));
typedef float v8f __attribute__((ext_vector_type(8)));
typedef float f4 __attribute__((ext_vector_type(4)));
typedef unsigned short us8 __attribute__((ext_vector_type(8)));
typedef unsigned short us4 __attribute__((ext_vector_type(4)));

union FragU { us8 h[2]; v16bf v; };

__device__ __forceinline__ unsigned short f2bf(float f) {
  // round-to-nearest-even f32 -> bf16
  unsigned int u = __float_as_uint(f);
  u += 0x7FFFu + ((u >> 16) & 1u);
  return (unsigned short)(u >> 16);
}

// A addressing modes
#define AM_ROW 0   // A[m*lda + k]          (row contiguous over k)
#define AM_COL 1   // A[k*lda + m]          (transA; contiguous over m)
// B addressing modes
#define BM_ROW 0   // B[k*ldb + n]          (contiguous over n)
#define BM_COLK 1  // B[n*ldb + k]          (transB; contiguous over k)
#define BM_CONV 2  // k' = c*3+tap : B[c*ldb + n + tap - 1], zero-padded cols

struct GemmP {
  const float* A; const float* B; float* C;
  const float* bias;     // per-M, optional
  const float* colmask;  // per-N (x_mask), optional
  long long sA, sB, sC, sMask;   // batch strides in elements
  int lda, ldb, ldc;
  int M, N, K;           // M%64==0, N%64==0, K%32==0 (no edge checks)
  float alpha;
};

// ---------------------------------------------------------------------------
// Specialized bf16 WMMA GEMM: C[M,N] = act(alpha * A*B + bias) [* colmask]
// Tile 64x64 (4 waves x 32x32), K step 32, double-buffered LDS with
// register-held staging: next tile's global_load_b128s are issued BEFORE the
// WMMAs on the current tile, so HBM weight streaming overlaps matrix math.
// One barrier per K step.  Fragments: two aligned ds_load_b128 each, matching
// the CDNA5 16-bit A 16x32 / B 32x16 VGPR layouts.
// ---------------------------------------------------------------------------
template <int AM, int BM, int GELU>
__global__ __launch_bounds__(128) void gemm_wmma(GemmP p) {
  __shared__ unsigned short Al[2][64][40]; // +8 pad; rows 80B => 16B aligned
  __shared__ unsigned short Bl[2][64][40];

  const int bz = blockIdx.z;
  const float* __restrict__ A = p.A + (long long)bz * p.sA;
  const float* __restrict__ Bm = p.B + (long long)bz * p.sB;
  float* __restrict__ Cm = p.C + (long long)bz * p.sC;

  const int m0 = blockIdx.y * 64;
  const int n0 = blockIdx.x * 64;
  const int tid = threadIdx.x;
  const int wave = tid >> 5, lane = tid & 31;
  const int wm = (wave >> 1) * 32;
  const int wn = (wave & 1) * 32;
  const int lrow = lane & 15, khalf = lane >> 4;

  const v8f vzero = {0.f,0.f,0.f,0.f,0.f,0.f,0.f,0.f};
  v8f acc[2][2];
  acc[0][0] = vzero; acc[0][1] = vzero; acc[1][0] = vzero; acc[1][1] = vzero;

  f4 ar[4], br[4];   // register-staged f32 tiles (in flight during WMMA)

  auto loadA = [&](int k0) {
    if (AM == AM_ROW) {
      for (int i = 0; i < 4; ++i) {
        int idx = tid + i * 128;
        int r = idx >> 3, cq = (idx & 7) * 4;
        ar[i] = *(const f4*)&A[(long long)(m0 + r) * p.lda + k0 + cq];
      }
    } else { // AM_COL: contiguous over m
      for (int i = 0; i < 4; ++i) {
        int idx = tid + i * 128;
        int c = idx >> 4, rq = (idx & 15) * 4;
        ar[i] = *(const f4*)&A[(long long)(k0 + c) * p.lda + m0 + rq];
      }
    }
  };
  auto storeA = [&](int buf) {
    if (AM == AM_ROW) {
      for (int i = 0; i < 4; ++i) {
        int idx = tid + i * 128;
        int r = idx >> 3, cq = (idx & 7) * 4;
        us4 w; w[0] = f2bf(ar[i][0]); w[1] = f2bf(ar[i][1]);
               w[2] = f2bf(ar[i][2]); w[3] = f2bf(ar[i][3]);
        *(us4*)&Al[buf][r][cq] = w;
      }
    } else {
      for (int i = 0; i < 4; ++i) {
        int idx = tid + i * 128;
        int c = idx >> 4, rq = (idx & 15) * 4;
        Al[buf][rq + 0][c] = f2bf(ar[i][0]); Al[buf][rq + 1][c] = f2bf(ar[i][1]);
        Al[buf][rq + 2][c] = f2bf(ar[i][2]); Al[buf][rq + 3][c] = f2bf(ar[i][3]);
      }
    }
  };
  auto loadB = [&](int k0) {
    if (BM == BM_ROW) {
      for (int i = 0; i < 4; ++i) {
        int idx = tid + i * 128;
        int c = idx >> 4, nq = (idx & 15) * 4;
        br[i] = *(const f4*)&Bm[(long long)(k0 + c) * p.ldb + n0 + nq];
      }
    } else if (BM == BM_COLK) {
      for (int i = 0; i < 4; ++i) {
        int idx = tid + i * 128;
        int r = idx >> 3, cq = (idx & 7) * 4;
        br[i] = *(const f4*)&Bm[(long long)(n0 + r) * p.ldb + k0 + cq];
      }
    } else { // BM_CONV
      for (int i = 0; i < 4; ++i) {
        int idx = tid + i * 128;
        int c = idx >> 4, nq = (idx & 15) * 4;
        int kk = k0 + c;
        int c3 = kk / 3;
        int tap = kk - c3 * 3;
        const float* row = Bm + (long long)c3 * p.ldb;
        int col0 = n0 + nq + tap - 1;
        f4 v;
        for (int j = 0; j < 4; ++j) {
          int col = col0 + j;
          v[j] = (col >= 0 && col < p.N) ? row[col] : 0.f;
        }
        br[i] = v;
      }
    }
  };
  auto storeB = [&](int buf) {
    if (BM == BM_COLK) {
      for (int i = 0; i < 4; ++i) {
        int idx = tid + i * 128;
        int r = idx >> 3, cq = (idx & 7) * 4;
        us4 w; w[0] = f2bf(br[i][0]); w[1] = f2bf(br[i][1]);
               w[2] = f2bf(br[i][2]); w[3] = f2bf(br[i][3]);
        *(us4*)&Bl[buf][r][cq] = w;
      }
    } else { // BM_ROW / BM_CONV: scatter along n
      for (int i = 0; i < 4; ++i) {
        int idx = tid + i * 128;
        int c = idx >> 4, nq = (idx & 15) * 4;
        Bl[buf][nq + 0][c] = f2bf(br[i][0]); Bl[buf][nq + 1][c] = f2bf(br[i][1]);
        Bl[buf][nq + 2][c] = f2bf(br[i][2]); Bl[buf][nq + 3][c] = f2bf(br[i][3]);
      }
    }
  };
  auto compute = [&](int buf) {
    for (int i = 0; i < 2; ++i) {
      FragU af;
      const int m = wm + i * 16 + lrow;
      af.h[0] = *(const us8*)&Al[buf][m][khalf * 8];        // K 0..7  / 8..15
      af.h[1] = *(const us8*)&Al[buf][m][16 + khalf * 8];   // K 16..23 / 24..31
      for (int j = 0; j < 2; ++j) {
        FragU bf;
        const int n = wn + j * 16 + lrow;
        bf.h[0] = *(const us8*)&Bl[buf][n][khalf * 8];      // K 0..7  / 8..15
        bf.h[1] = *(const us8*)&Bl[buf][n][16 + khalf * 8]; // K 16..23 / 24..31
        acc[i][j] = __builtin_amdgcn_wmma_f32_16x16x32_bf16(
            false, af.v, false, bf.v, (short)0, acc[i][j], false, false);
      }
    }
  };

  // ---- software pipeline: prologue ----
  loadA(0); loadB(0);
  storeA(0); storeB(0);
  __syncthreads();

  int buf = 0;
  for (int k0 = 0;;) {
    const bool last = (k0 + 32 >= p.K);
    if (!last) {
      loadA(k0 + 32);           // global loads issued BEFORE the WMMAs...
      loadB(k0 + 32);
      if (k0 + 64 < p.K) {      // ...and prefetch one more step of weights
        if (AM == AM_ROW)
          __builtin_prefetch(A + (long long)(m0 + (tid >> 1)) * p.lda + k0 + 64 + (tid & 1) * 16, 0, 3);
        else
          __builtin_prefetch(A + (long long)(k0 + 64 + (tid >> 2)) * p.lda + m0 + (tid & 3) * 16, 0, 3);
      }
    }
    compute(buf);               // matrix math overlaps the in-flight loads
    if (last) break;
    storeA(buf ^ 1); storeB(buf ^ 1);
    __syncthreads();            // one barrier per K step
    k0 += 32; buf ^= 1;
  }

  // ---- epilogue: alpha, bias, gelu, column mask (no edge checks needed) ----
  for (int i = 0; i < 2; ++i)
    for (int j = 0; j < 2; ++j)
      for (int e = 0; e < 8; ++e) {
        int m = m0 + wm + i * 16 + khalf * 8 + e;
        int n = n0 + wn + j * 16 + lrow;
        float v = acc[i][j][e] * p.alpha;
        if (p.bias) v += p.bias[m];
        if (GELU) v = 0.5f * v * (1.f + erff(v * 0.70710678f));
        if (p.colmask) v *= p.colmask[(long long)bz * p.sMask + n];
        Cm[(long long)m * p.ldc + n] = v;
      }
}

// ---------------------------------------------------------------------------
// Softmax over s with fused banded relative-position logits and mask.
// One 256-thread block per (b,h,t) row; scores updated in place.
// rel term nonzero only for |s-t| <= W=4 (emb width 9 after padding algebra).
// ---------------------------------------------------------------------------
__global__ __launch_bounds__(256) void softmax_rel(
    float* __restrict__ sc, const float* __restrict__ q,
    const float* __restrict__ ek, const float* __restrict__ xmask) {
  __shared__ float rb[9];
  __shared__ float red[256];
  const int idx = blockIdx.x;
  const int t = idx % Tt;
  const int bh = idx / Tt;
  const int h = bh % Hh, b = bh / Hh;
  const int tid = threadIdx.x;

  if (tid < 9) {
    float s = 0.f;
    const float* qcol = q + ((long long)b * Cc + h * Dd) * Tt + t;
    for (int d = 0; d < Dd; ++d) s += qcol[(long long)d * Tt] * ek[d * 9 + tid];
    rb[tid] = 0.125f * s;  // (q / sqrt(D)) . k_emb
  }
  __syncthreads();

  float* row = sc + ((long long)bh * Tt + t) * Tt;
  const float mt = xmask[b * Tt + t];
  const int s0 = tid, s1 = tid + 256;

  auto getv = [&](int s) {
    float v = row[s];
    int dt = s - t;
    if (dt >= -4 && dt <= 4) v += rb[dt + 4];
    if (mt * xmask[b * Tt + s] == 0.f) v = -10000.f;
    return v;
  };
  float v0 = getv(s0), v1 = getv(s1);

  red[tid] = fmaxf(v0, v1);
  __syncthreads();
  for (int off = 128; off > 0; off >>= 1) {
    if (tid < off) red[tid] = fmaxf(red[tid], red[tid + off]);
    __syncthreads();
  }
  const float mx = red[0];
  __syncthreads();

  const float e0 = expf(v0 - mx), e1 = expf(v1 - mx);
  red[tid] = e0 + e1;
  __syncthreads();
  for (int off = 128; off > 0; off >>= 1) {
    if (tid < off) red[tid] += red[tid + off];
    __syncthreads();
  }
  const float inv = 1.f / red[0];
  row[s0] = e0 * inv;
  row[s1] = e1 * inv;
}

// out[b, h*D+d, t] += sum_{j=0..8} p[b,h,t,t+j-4] * ev[d,j]
__global__ __launch_bounds__(256) void add_rel_v(
    float* __restrict__ attnout, const float* __restrict__ p,
    const float* __restrict__ ev) {
  const long long g = (long long)blockIdx.x * 256 + threadIdx.x;
  const int t = (int)(g % Tt);
  const int c = (int)((g / Tt) % Cc);
  const int b = (int)(g / ((long long)Cc * Tt));
  const int h = c >> 6, d = c & 63;
  const float* prow = p + (((long long)(b * Hh + h) * Tt) + t) * Tt;
  float s = attnout[g];
  for (int j = 0; j < 9; ++j) {
    int sc = t + j - 4;
    if (sc >= 0 && sc < Tt) s += prow[sc] * ev[d * 9 + j];
  }
  attnout[g] = s;
}

// channel LayerNorm of (x + y), optional final column-mask multiply
__global__ __launch_bounds__(256) void ln_res(
    const float* __restrict__ x, const float* __restrict__ y,
    float* __restrict__ out, const float* __restrict__ gam,
    const float* __restrict__ bet, const float* __restrict__ xmask,
    int useMask) {
  __shared__ float red[256];
  const int bt = blockIdx.x;
  const int b = bt / Tt, t = bt % Tt;
  const int tid = threadIdx.x;
  const long long base = (long long)b * Cc * Tt + t;

  float v[3];
  for (int i = 0; i < 3; ++i) {
    int c = tid + i * 256;
    v[i] = x[base + (long long)c * Tt] + y[base + (long long)c * Tt];
  }
  red[tid] = v[0] + v[1] + v[2];
  __syncthreads();
  for (int off = 128; off > 0; off >>= 1) {
    if (tid < off) red[tid] += red[tid + off];
    __syncthreads();
  }
  const float mean = red[0] * (1.f / (float)Cc);
  __syncthreads();
  float sq = 0.f;
  for (int i = 0; i < 3; ++i) { float d = v[i] - mean; sq += d * d; }
  red[tid] = sq;
  __syncthreads();
  for (int off = 128; off > 0; off >>= 1) {
    if (tid < off) red[tid] += red[tid + off];
    __syncthreads();
  }
  const float inv = rsqrtf(red[0] * (1.f / (float)Cc) + 1e-5f);
  const float msk = useMask ? xmask[b * Tt + t] : 1.f;
  for (int i = 0; i < 3; ++i) {
    int c = tid + i * 256;
    out[base + (long long)c * Tt] = ((v[i] - mean) * inv * gam[c] + bet[c]) * msk;
  }
}

// ---------------------------------------------------------------------------
static inline GemmP mk(const float* A, const float* Bm, float* Cm,
                       int M, int N, int K, int lda, int ldb, int ldc,
                       long long sA, long long sB, long long sC,
                       float alpha, const float* bias) {
  GemmP p{};
  p.A = A; p.B = Bm; p.C = Cm; p.bias = bias; p.colmask = nullptr;
  p.sA = sA; p.sB = sB; p.sC = sC; p.sMask = 0;
  p.lda = lda; p.ldb = ldb; p.ldc = ldc;
  p.M = M; p.N = N; p.K = K;
  p.alpha = alpha;
  return p;
}

template <int AM, int BM, int GELU>
static inline void run_gemm(const GemmP& p, int batch, hipStream_t s) {
  dim3 grid((unsigned)(p.N / 64), (unsigned)(p.M / 64), (unsigned)batch);
  gemm_wmma<AM, BM, GELU><<<grid, 128, 0, s>>>(p);
}

extern "C" void kernel_launch(void* const* d_in, const int* in_sizes, int n_in,
                              void* d_out, int out_size, void* d_ws, size_t ws_size,
                              hipStream_t stream) {
  (void)in_sizes; (void)n_in; (void)out_size; (void)ws_size;
  const float* x     = (const float*)d_in[0];
  const float* xmask = (const float*)d_in[1];
  const float* Wq = (const float*)d_in[2];  const float* bq = (const float*)d_in[3];
  const float* Wk = (const float*)d_in[4];  const float* bk = (const float*)d_in[5];
  const float* Wv = (const float*)d_in[6];  const float* bv = (const float*)d_in[7];
  const float* Wo = (const float*)d_in[8];  const float* bo = (const float*)d_in[9];
  const float* ek = (const float*)d_in[10]; const float* ev = (const float*)d_in[11];
  const float* w1 = (const float*)d_in[12]; const float* b1 = (const float*)d_in[13];
  const float* w2 = (const float*)d_in[14]; const float* b2 = (const float*)d_in[15];
  const float* g1 = (const float*)d_in[16]; const float* be1 = (const float*)d_in[17];
  const float* g2 = (const float*)d_in[18]; const float* be2 = (const float*)d_in[19];

  const long long CT = (long long)Cc * Tt;      // 393216
  const long long DT = (long long)Dd * Tt;      // 32768
  const long long TT = (long long)Tt * Tt;      // 262144
  const long long BCT = (long long)Bb * CT;

  float* ws = (float*)d_ws;
  float* qb  = ws;              // reused as projection/ffn output 'y' after softmax
  float* kb  = ws + BCT;
  float* vb  = ws + 2 * BCT;
  float* ao  = ws + 3 * BCT;    // attention output (pre-projection)
  float* xb  = ws + 4 * BCT;    // running x
  float* scb = ws + 5 * BCT;    // scores / softmax P ; reused as FFN hidden
  float* yb  = qb;

  const float* curx = x;
  for (int l = 0; l < NLl; ++l) {
    const float* Wq_l = Wq + (long long)l * Cc * Cc; const float* bq_l = bq + l * Cc;
    const float* Wk_l = Wk + (long long)l * Cc * Cc; const float* bk_l = bk + l * Cc;
    const float* Wv_l = Wv + (long long)l * Cc * Cc; const float* bv_l = bv + l * Cc;
    const float* Wo_l = Wo + (long long)l * Cc * Cc; const float* bo_l = bo + l * Cc;
    const float* ek_l = ek + (long long)l * Dd * 9;  const float* ev_l = ev + (long long)l * Dd * 9;
    const float* w1_l = w1 + (long long)l * FFf * Cc * 3; const float* b1_l = b1 + l * FFf;
    const float* w2_l = w2 + (long long)l * Cc * FFf * 3; const float* b2_l = b2 + l * Cc;

    // Q, K, V projections: [C,C] x [C,T] per batch
    run_gemm<AM_ROW, BM_ROW, 0>(
        mk(Wq_l, curx, qb, Cc, Tt, Cc, Cc, Tt, Tt, 0, CT, CT, 1.f, bq_l), Bb, stream);
    run_gemm<AM_ROW, BM_ROW, 0>(
        mk(Wk_l, curx, kb, Cc, Tt, Cc, Cc, Tt, Tt, 0, CT, CT, 1.f, bk_l), Bb, stream);
    run_gemm<AM_ROW, BM_ROW, 0>(
        mk(Wv_l, curx, vb, Cc, Tt, Cc, Cc, Tt, Tt, 0, CT, CT, 1.f, bv_l), Bb, stream);

    // scores[t,s] = (1/8) * sum_d q[d,t] k[d,s]   (A = q transposed)
    run_gemm<AM_COL, BM_ROW, 0>(
        mk(qb, kb, scb, Tt, Tt, Dd, Tt, Tt, Tt, DT, DT, TT, 0.125f, nullptr),
        Bb * Hh, stream);

    // fused banded rel-logits + mask + softmax (in place)
    softmax_rel<<<Bb * Hh * Tt, 256, 0, stream>>>(scb, qb, ek_l, xmask);

    // out[d,t] = sum_s v[d,s] p[t,s]   (B = P transposed)
    run_gemm<AM_ROW, BM_COLK, 0>(
        mk(vb, scb, ao, Dd, Tt, Tt, Tt, Tt, Tt, DT, TT, DT, 1.f, nullptr),
        Bb * Hh, stream);

    // banded value-side relative term
    add_rel_v<<<(int)(BCT / 256), 256, 0, stream>>>(ao, scb, ev_l);

    // output projection (q buffer now free -> y)
    run_gemm<AM_ROW, BM_ROW, 0>(
        mk(Wo_l, ao, yb, Cc, Tt, Cc, Cc, Tt, Tt, 0, CT, CT, 1.f, bo_l), Bb, stream);

    // x = LN(x + y)
    ln_res<<<Bb * Tt, 256, 0, stream>>>(curx, yb, xb, g1 + l * Cc, be1 + l * Cc, xmask, 0);

    // FFN conv1 (K=3, pad=1) as one GEMM with K' = 3*C (im2col on B side),
    // + GELU + column mask -> hidden (reuses scores buf)
    GemmP p1 = mk(w1_l, xb, scb, FFf, Tt, Cc * 3, Cc * 3, Tt, Tt,
                  0, CT, (long long)FFf * Tt, 1.f, b1_l);
    p1.colmask = xmask; p1.sMask = Tt;
    run_gemm<AM_ROW, BM_CONV, 1>(p1, Bb, stream);

    // FFN conv2: K' = 3*4C -> y
    GemmP p2 = mk(w2_l, scb, yb, Cc, Tt, FFf * 3, FFf * 3, Tt, Tt,
                  0, (long long)FFf * Tt, CT, 1.f, b2_l);
    run_gemm<AM_ROW, BM_CONV, 0>(p2, Bb, stream);

    // x = LN(x + y) * mask  (mask fold is exact: masked columns end up zero)
    float* outp = (l == NLl - 1) ? (float*)d_out : xb;
    ln_res<<<Bb * Tt, 256, 0, stream>>>(xb, yb, outp, g2 + l * Cc, be2 + l * Cc, xmask, 1);

    curx = xb;
  }
}
